// Net_53798760349963
// MI455X (gfx1250) — matrix operand order repaired and checked by reading, not compiled, not run on hardware
//
#include <hip/hip_runtime.h>
#include <hip/hip_bf16.h>

typedef __attribute__((ext_vector_type(16))) _Float16 v16h;
typedef __attribute__((ext_vector_type(8)))  float    v8f;

// ---------------------------------------------------------------------------
// Weight packing: W[K,N] f32 row-major -> f16 WMMA B-fragments.
// Layout: frag index ((tn*Kc + kc)*32 + lane) * 16 halves.
// Contiguous k order, not the ISA table order (16-bit B, 32x16): lanes 0-15 hold K = 0..15 of the chunk
// (2 per VGPR), lanes 16-31 hold K = 16..31.  N column = tn*16 + (lane&15).
// ---------------------------------------------------------------------------
__global__ void packB(const float* __restrict__ W, _Float16* __restrict__ out,
                      int K, int N) {
    int tid = blockIdx.x * blockDim.x + threadIdx.x;
    int Kc = K >> 5, Ntiles = N >> 4;
    int total = Ntiles * Kc * 32;
    if (tid >= total) return;
    int lane = tid & 31;
    int rest = tid >> 5;
    int kc = rest % Kc, tn = rest / Kc;
    int kb = (lane >> 4) << 4;            // 0 or 16
    int n  = (tn << 4) + (lane & 15);
    int k0 = (kc << 5) + kb;
    _Float16* o = out + (size_t)tid * 16;
#pragma unroll
    for (int i = 0; i < 8; ++i) {
        o[2*i]   = (_Float16)W[(size_t)(k0 + 2*i)     * N + n];
        o[2*i+1] = (_Float16)W[(size_t)(k0 + 2*i + 1) * N + n];
    }
}

// ---------------------------------------------------------------------------
// C[M,N] = act(A[M,K] @ B[K,N] + bias).  A f32 row-major (lda == K),
// B pre-packed f16 fragments, C f32 row-major.
// One 16x64 strip (4 N-tiles) per wave: A fragment loaded/converted once per
// K-chunk and reused across 4 v_wmma issues (4x A-traffic reduction).
// Requires M%16==0, K%32==0, N%64==0 (true for every call site).
// ---------------------------------------------------------------------------
template <int ACT>
__global__ __launch_bounds__(256) void gemm_wmma(
    const float* __restrict__ A, const _Float16* __restrict__ Bp,
    const float* __restrict__ bias, float* __restrict__ C,
    int M, int N, int K) {
    const int Mtiles = M >> 4, Ntiles = N >> 4, Kc = K >> 5;
    const int Ngrp = Ntiles >> 2;                       // groups of 4 N-tiles
    const int wave = blockIdx.x * (blockDim.x >> 5) + (threadIdx.x >> 5);
    if (wave >= Mtiles * Ngrp) return;
    const int lane = threadIdx.x & 31;
    const int tm = wave / Ngrp, tg = wave % Ngrp;
    const int half = lane >> 4, l16 = lane & 15;
    // A fragment: lane row m = l16; K set {ka..ka+15} (contiguous, matches packB)
    const int am = (tm << 4) + l16;
    const int ka = half << 4;
    const size_t arow = (size_t)am * K;
    // B fragment bases for the 4 consecutive N-tiles
    const size_t tstride = (size_t)Kc * 512;            // halves per N-tile
    const _Float16* b0p = Bp + (size_t)(tg * 4 + 0) * tstride + lane * 16;
    const _Float16* b1p = b0p + tstride;
    const _Float16* b2p = b1p + tstride;
    const _Float16* b3p = b2p + tstride;

    v8f acc0 = {}, acc1 = {}, acc2 = {}, acc3 = {};
    for (int kc = 0; kc < Kc; ++kc) {
        const int k0 = kc << 5;
        const float4* ap = (const float4*)(A + arow + k0 + ka);
        float4 a0 = ap[0], a1 = ap[1], a2 = ap[2], a3 = ap[3];
        v16h a;
        a[0]=(_Float16)a0.x;  a[1]=(_Float16)a0.y;  a[2]=(_Float16)a0.z;  a[3]=(_Float16)a0.w;
        a[4]=(_Float16)a1.x;  a[5]=(_Float16)a1.y;  a[6]=(_Float16)a1.z;  a[7]=(_Float16)a1.w;
        a[8]=(_Float16)a2.x;  a[9]=(_Float16)a2.y;  a[10]=(_Float16)a2.z; a[11]=(_Float16)a2.w;
        a[12]=(_Float16)a3.x; a[13]=(_Float16)a3.y; a[14]=(_Float16)a3.z; a[15]=(_Float16)a3.w;
        const size_t ko = (size_t)kc * 512;
        v16h b0 = *(const v16h*)(b0p + ko);
        v16h b1 = *(const v16h*)(b1p + ko);
        v16h b2 = *(const v16h*)(b2p + ko);
        v16h b3 = *(const v16h*)(b3p + ko);
        acc0 = __builtin_amdgcn_wmma_f32_16x16x32_f16(false, a, false, b0, (short)0, acc0, false, false);
        acc1 = __builtin_amdgcn_wmma_f32_16x16x32_f16(false, a, false, b1, (short)0, acc1, false, false);
        acc2 = __builtin_amdgcn_wmma_f32_16x16x32_f16(false, a, false, b2, (short)0, acc2, false, false);
        acc3 = __builtin_amdgcn_wmma_f32_16x16x32_f16(false, a, false, b3, (short)0, acc3, false, false);
    }
    const int rbase = (tm << 4) + (half << 3);
#pragma unroll
    for (int j = 0; j < 4; ++j) {
        const v8f acc = (j == 0) ? acc0 : (j == 1) ? acc1 : (j == 2) ? acc2 : acc3;
        const int col = ((tg * 4 + j) << 4) + l16;
        const float bv = bias ? bias[col] : 0.f;
#pragma unroll
        for (int r = 0; r < 8; ++r) {
            float v = acc[r] + bv;
            if (ACT == 1) v = v > 0.f ? v : 0.f;
            C[(size_t)(rbase + r) * N + col] = v;
        }
    }
}

// ---------------------------------------------------------------------------
// Elementwise / gather / scatter / segment kernels
// ---------------------------------------------------------------------------
__global__ void fillf(float* p, float v, int n) {
    int i = blockIdx.x * blockDim.x + threadIdx.x;
    if (i < n) p[i] = v;
}

__global__ void proj5(const float* __restrict__ X, const int* __restrict__ idx,
                      int useIdx, const float* __restrict__ W,
                      const float* __restrict__ b, float* __restrict__ Y,
                      int nrows, int C) {
    int i = blockIdx.x * blockDim.x + threadIdx.x;
    if (i >= nrows * C) return;
    int r = i / C, c = i % C;
    int sr = useIdx ? (idx[r] - 1) : r;
    const float* x = X + (size_t)sr * 5;
    float acc = b[c];
#pragma unroll
    for (int k = 0; k < 5; ++k) acc += x[k] * W[k * C + c];
    Y[i] = acc;
}

__global__ void gather_rows(const float* __restrict__ state,
                            const int* __restrict__ idx, float* __restrict__ Y,
                            int n) {
    int i = blockIdx.x * blockDim.x + threadIdx.x;
    if (i >= n * 128) return;
    int r = i >> 7, j = i & 127;
    Y[i] = state[(size_t)idx[r] * 128 + j];
}

__global__ void scatter_rows(float* __restrict__ state,
                             const int* __restrict__ idx,
                             const float* __restrict__ Y, int n) {
    int i = blockIdx.x * blockDim.x + threadIdx.x;
    if (i >= n * 128) return;
    int r = i >> 7, j = i & 127;
    state[(size_t)idx[r] * 128 + j] = Y[i];
}

__global__ void gather_xt(const float* __restrict__ state,
                          const int* __restrict__ exeSeqs,
                          float* __restrict__ xt, int t, int S, int L) {
    int i = blockIdx.x * blockDim.x + threadIdx.x;
    if (i >= S * 256) return;
    int s = i >> 8, j = i & 255;
    int node = (j < 128) ? exeSeqs[(size_t)s * 2 * L + 2 * t]
                         : exeSeqs[(size_t)s * 2 * L + 2 * t + 1];
    xt[i] = state[(size_t)node * 128 + (j & 127)];
}

__global__ void gru_pointwise(const float* __restrict__ gi,
                              const float* __restrict__ gh,
                              const float* __restrict__ hin,
                              float* __restrict__ hout, float* __restrict__ outs,
                              const int* __restrict__ lengths, int t,
                              int nrows, int H, int L) {
    int i = blockIdx.x * blockDim.x + threadIdx.x;
    if (i >= nrows * H) return;
    int row = i / H, j = i % H;
    const float* gir = gi + (size_t)row * 3 * H;
    const float* ghr = gh + (size_t)row * 3 * H;
    float ir = gir[j], iz = gir[H + j], ig = gir[2 * H + j];
    float hr = ghr[j], hz = ghr[H + j], hg = ghr[2 * H + j];
    float r = 1.f / (1.f + __expf(-(ir + hr)));
    float z = 1.f / (1.f + __expf(-(iz + hz)));
    float n = tanhf(ig + r * hg);
    float hv = hin[(size_t)row * H + j];
    float hn = (1.f - z) * n + z * hv;
    if (lengths && t >= lengths[row]) hn = hv;   // packed-sequence freeze
    hout[(size_t)row * H + j] = hn;
    if (outs) outs[((size_t)row * L + t) * H + j] = hn;
}

__global__ void gather_outf(const float* __restrict__ outs,
                            const int* __restrict__ fsvc,
                            const int* __restrict__ fpos,
                            float* __restrict__ out_f, int F, int L) {
    int i = blockIdx.x * blockDim.x + threadIdx.x;
    if (i >= F * 128) return;
    int f = i >> 7, j = i & 127;
    out_f[i] = outs[(((size_t)fsvc[f] * L) + fpos[f]) * 128 + j];
}

// out[f] = concat(left[f], right[idx? idx[f] : f])
__global__ void concat128(const float* __restrict__ left,
                          const float* __restrict__ right,
                          const int* __restrict__ idx, float* __restrict__ out,
                          int n) {
    int i = blockIdx.x * blockDim.x + threadIdx.x;
    if (i >= n * 256) return;
    int f = i >> 8, j = i & 255;
    if (j < 128) out[i] = left[(size_t)f * 128 + j];
    else {
        int rr = idx ? idx[f] : f;
        out[i] = right[(size_t)rr * 128 + (j - 128)];
    }
}

__device__ inline void atomicMaxF(float* addr, float val) {
    if (val >= 0.f) atomicMax((int*)addr, __float_as_int(val));
    else atomicMin((unsigned int*)addr, __float_as_uint(val));
}

__global__ void gat_alpha(const float* __restrict__ src,
                          const float* __restrict__ tgt,
                          const int* __restrict__ dev,
                          const float* __restrict__ att,
                          float* __restrict__ alpha, float* __restrict__ amax) {
    __shared__ float red[256];
    int f = blockIdx.x, tid = threadIdx.x;
    int d = dev[f];
    for (int h = 0; h < 4; ++h) {
        float e = src[(size_t)f * 1024 + h * 256 + tid] +
                  tgt[(size_t)d * 1024 + h * 256 + tid];
        e = (e > 0.f) ? e : 0.2f * e;                 // leaky relu
        red[tid] = e * att[h * 256 + tid];
        __syncthreads();
        for (int s = 128; s > 0; s >>= 1) {
            if (tid < s) red[tid] += red[tid + s];
            __syncthreads();
        }
        if (tid == 0) {
            float a = red[0];
            alpha[f * 4 + h] = a;
            atomicMaxF(&amax[d * 4 + h], a);
        }
        __syncthreads();
    }
}

__global__ void gat_w(const float* __restrict__ alpha,
                      const float* __restrict__ amax,
                      const int* __restrict__ dev, float* __restrict__ w,
                      float* __restrict__ wsum, int F) {
    int i = blockIdx.x * blockDim.x + threadIdx.x;
    if (i >= F * 4) return;
    int f = i >> 2, h = i & 3;
    int d = dev[f];
    float wv = __expf(alpha[i] - amax[d * 4 + h]);
    w[i] = wv;
    atomicAdd(&wsum[d * 4 + h], wv);
}

__global__ void gat_msgd(const float* __restrict__ src,
                         const float* __restrict__ w,
                         const float* __restrict__ wsum,
                         const int* __restrict__ dev, float* __restrict__ msgD,
                         int F) {
    int i = blockIdx.x * blockDim.x + threadIdx.x;
    if (i >= F * 1024) return;
    int f = i >> 10, c = i & 1023, h = c >> 8;
    int d = dev[f];
    float a = w[f * 4 + h] / (wsum[d * 4 + h] + 1e-16f);
    atomicAdd(&msgD[(size_t)d * 1024 + c], src[i] * a);
}

__global__ void gat_mean(const float* __restrict__ msgD,
                         const float* __restrict__ bias,
                         float* __restrict__ out, int Dn) {
    int i = blockIdx.x * blockDim.x + threadIdx.x;
    if (i >= Dn * 256) return;
    int d = i >> 8, c = i & 255;
    const float* m = msgD + (size_t)d * 1024;
    out[i] = 0.25f * (m[c] + m[256 + c] + m[512 + c] + m[768 + c]) + bias[c];
}

__global__ void pool_acc(const float* __restrict__ frag,
                         const int* __restrict__ fsvc, float* __restrict__ pooled,
                         int F) {
    int i = blockIdx.x * blockDim.x + threadIdx.x;
    if (i >= F * 128) return;
    int f = i >> 7, j = i & 127;
    atomicAdd(&pooled[(size_t)fsvc[f] * 128 + j], frag[i]);
}

__global__ void pool_scale(float* __restrict__ pooled,
                           const int* __restrict__ lengths, int S) {
    int i = blockIdx.x * blockDim.x + threadIdx.x;
    if (i >= S * 128) return;
    pooled[i] /= (float)lengths[i >> 7];
}

__global__ void dot_out(const float* __restrict__ H, const float* __restrict__ W3,
                        const float* __restrict__ b3, float* __restrict__ out,
                        int Kdim) {
    __shared__ float red[256];
    int s = blockIdx.x, tid = threadIdx.x;
    red[tid] = (tid < Kdim) ? H[(size_t)s * Kdim + tid] * W3[tid] : 0.f;
    __syncthreads();
    for (int st = 128; st > 0; st >>= 1) {
        if (tid < st) red[tid] += red[tid + st];
        __syncthreads();
    }
    if (tid == 0) out[s] = red[0] + b3[0];
}

// ---------------------------------------------------------------------------
// Host orchestration
// ---------------------------------------------------------------------------
extern "C" void kernel_launch(void* const* d_in, const int* in_sizes, int n_in,
                              void* d_out, int out_size, void* d_ws, size_t ws_size,
                              hipStream_t stream) {
    const float* realnode = (const float*)d_in[0];
    const float* arr      = (const float*)d_in[1];
    const int* exeSeqs        = (const int*)d_in[2];
    const int* index_devices  = (const int*)d_in[3];
    const int* index_fragments= (const int*)d_in[4];
    const int* frag2dev_node  = (const int*)d_in[5];
    const int* frag2dev_row   = (const int*)d_in[6];
    const int* lengths        = (const int*)d_in[7];
    const int* frag_service   = (const int*)d_in[8];
    const int* frag_pos       = (const int*)d_in[9];
    const float *W_dev=(const float*)d_in[10], *b_dev=(const float*)d_in[11];
    const float *W_frg=(const float*)d_in[12], *b_frg=(const float*)d_in[13];
    const float *W_srv=(const float*)d_in[14], *b_srv=(const float*)d_in[15];
    const float *C_Wih=(const float*)d_in[16], *C_Whh=(const float*)d_in[17];
    const float *C_bih=(const float*)d_in[18], *C_bhh=(const float*)d_in[19];
    const float *F_Wih=(const float*)d_in[20], *F_Whh=(const float*)d_in[21];
    const float *F_bih=(const float*)d_in[22], *F_bhh=(const float*)d_in[23];
    const float *D_Wih=(const float*)d_in[24], *D_Whh=(const float*)d_in[25];
    const float *D_bih=(const float*)d_in[26], *D_bhh=(const float*)d_in[27];
    const float *W_src=(const float*)d_in[28], *b_src=(const float*)d_in[29];
    const float *W_tgt=(const float*)d_in[30], *b_tgt=(const float*)d_in[31];
    const float *att=(const float*)d_in[32],   *bias_att=(const float*)d_in[33];
    const float *Wt1=(const float*)d_in[34], *bt1=(const float*)d_in[35];
    const float *Wt2=(const float*)d_in[36], *bt2=(const float*)d_in[37];
    const float *Wt3=(const float*)d_in[38], *bt3=(const float*)d_in[39];
    const float *Wl1=(const float*)d_in[40], *bl1=(const float*)d_in[41];
    const float *Wl2=(const float*)d_in[42], *bl2=(const float*)d_in[43];
    const float *Wl3=(const float*)d_in[44], *bl3=(const float*)d_in[45];

    const int S  = in_sizes[7];              // 2048
    const int F  = in_sizes[4];              // 20480
    const int Dn = in_sizes[3];              // 4096
    const int L  = in_sizes[2] / (2 * S);    // 16
    const int nodes = in_sizes[0] / 5 + 1;   // 24577

    // ---- workspace bump allocator ----
    char* base = (char*)d_ws;
    size_t off = 0;
    auto allocf = [&](size_t n) -> float* {
        float* p = (float*)(base + off);
        off = (off + n * sizeof(float) + 255) & ~(size_t)255;
        return p;
    };
    auto alloch = [&](size_t n) -> _Float16* {
        _Float16* p = (_Float16*)(base + off);
        off = (off + n * sizeof(_Float16) + 255) & ~(size_t)255;
        return p;
    };
    float* state    = allocf((size_t)nodes * 128);
    float* h_srv    = allocf((size_t)S * 128);
    float* outs     = allocf((size_t)S * L * 128);
    float* x_t      = allocf((size_t)S * 256);
    float* gi       = allocf((size_t)F * 384);   // reused by all 3 GRUs
    float* gh       = allocf((size_t)F * 384);
    float* out_f    = allocf((size_t)F * 128);
    float* msgF     = allocf((size_t)F * 256);
    float* srcin    = allocf((size_t)F * 256);
    float* frag_cur = allocf((size_t)F * 128);
    float* frag_new = allocf((size_t)F * 128);
    float* dev_cur  = allocf((size_t)Dn * 128);
    float* dev_new  = allocf((size_t)Dn * 128);
    float* srcb     = allocf((size_t)F * 1024);
    float* tgtb     = allocf((size_t)Dn * 1024);
    float* alphab   = allocf((size_t)F * 4);
    float* wbuf     = allocf((size_t)F * 4);
    float* amax     = allocf((size_t)Dn * 4);
    float* wsum     = allocf((size_t)Dn * 4);
    float* msgD     = allocf((size_t)Dn * 1024);
    float* msgDm    = allocf((size_t)Dn * 256);
    float* pooled   = allocf((size_t)S * 128);
    float* mh1      = allocf((size_t)S * 256);
    float* mh2      = allocf((size_t)S * 256);
    _Float16* pCWih = alloch(256 * 384);
    _Float16* pCWhh = alloch(128 * 384);
    _Float16* pFWih = alloch(256 * 384);
    _Float16* pFWhh = alloch(128 * 384);
    _Float16* pDWih = alloch(256 * 384);
    _Float16* pDWhh = alloch(128 * 384);
    _Float16* pWsrc = alloch(256 * 1024);
    _Float16* pWtgt = alloch(128 * 1024);
    _Float16* pWt1  = alloch(128 * 256);
    _Float16* pWt2  = alloch(256 * 256);
    _Float16* pWl1  = alloch(128 * 256);
    _Float16* pWl2  = alloch(256 * 256);
    float* dout = (float*)d_out;

    auto blk = [](long long n) { return (int)((n + 255) / 256); };
    auto pack = [&](const float* W, _Float16* o, int K, int N) {
        int th = (N / 16) * (K / 32) * 32;
        packB<<<blk(th), 256, 0, stream>>>(W, o, K, N);
    };
    auto gemm = [&](const float* A, const _Float16* Bp, const float* bias,
                    float* C, int M, int N, int K, int act) {
        int waves = (M / 16) * (N / 64);    // one 16x64 strip per wave
        int blocks = (waves + 7) / 8;       // 8 waves (256 threads) per block
        if (act) gemm_wmma<1><<<blocks, 256, 0, stream>>>(A, Bp, bias, C, M, N, K);
        else     gemm_wmma<0><<<blocks, 256, 0, stream>>>(A, Bp, bias, C, M, N, K);
    };

    // ---- pack weights (every call; no cached state) ----
    pack(C_Wih, pCWih, 256, 384);  pack(C_Whh, pCWhh, 128, 384);
    pack(F_Wih, pFWih, 256, 384);  pack(F_Whh, pFWhh, 128, 384);
    pack(D_Wih, pDWih, 256, 384);  pack(D_Whh, pDWhh, 128, 384);
    pack(W_src, pWsrc, 256, 1024); pack(W_tgt, pWtgt, 128, 1024);
    pack(Wt1, pWt1, 128, 256);     pack(Wt2, pWt2, 256, 256);
    pack(Wl1, pWl1, 128, 256);     pack(Wl2, pWl2, 256, 256);

    // ---- initial node states ----
    fillf<<<blk((long long)nodes * 128), 256, 0, stream>>>(state, 0.f, nodes * 128);
    proj5<<<blk((long long)Dn * 128), 256, 0, stream>>>(realnode, index_devices, 1,
                                                        W_dev, b_dev, dev_new, Dn, 128);
    proj5<<<blk((long long)F * 128), 256, 0, stream>>>(realnode, index_fragments, 1,
                                                       W_frg, b_frg, frag_new, F, 128);
    scatter_rows<<<blk((long long)Dn * 128), 256, 0, stream>>>(state, index_devices, dev_new, Dn);
    scatter_rows<<<blk((long long)F * 128), 256, 0, stream>>>(state, index_fragments, frag_new, F);
    proj5<<<blk((long long)S * 128), 256, 0, stream>>>(arr, nullptr, 0, W_srv, b_srv,
                                                       h_srv, S, 128);

    for (int iter = 0; iter < 3; ++iter) {
        // --- phiC: GRU scan over execution sequences ---
        for (int t = 0; t < L; ++t) {
            gather_xt<<<blk((long long)S * 256), 256, 0, stream>>>(state, exeSeqs, x_t, t, S, L);
            gemm(x_t, pCWih, C_bih, gi, S, 384, 256, 0);
            gemm(h_srv, pCWhh, C_bhh, gh, S, 384, 128, 0);
            gru_pointwise<<<blk((long long)S * 128), 256, 0, stream>>>(
                gi, gh, h_srv, h_srv, outs, lengths, t, S, 128, L);
        }
        gather_outf<<<blk((long long)F * 128), 256, 0, stream>>>(outs, frag_service,
                                                                 frag_pos, out_f, F, L);
        gather_rows<<<blk((long long)F * 128), 256, 0, stream>>>(state, index_fragments,
                                                                 frag_cur, F);
        // --- fragment GRU ---
        concat128<<<blk((long long)F * 256), 256, 0, stream>>>(out_f, state,
                                                               frag2dev_node, msgF, F);
        gemm(msgF, pFWih, F_bih, gi, F, 384, 256, 0);
        gemm(frag_cur, pFWhh, F_bhh, gh, F, 384, 128, 0);
        gru_pointwise<<<blk((long long)F * 128), 256, 0, stream>>>(
            gi, gh, frag_cur, frag_new, nullptr, nullptr, 0, F, 128, 0);
        // --- GAT scatter-softmax ---
        concat128<<<blk((long long)F * 256), 256, 0, stream>>>(out_f, frag_cur,
                                                               nullptr, srcin, F);
        gemm(srcin, pWsrc, b_src, srcb, F, 1024, 256, 0);
        gather_rows<<<blk((long long)Dn * 128), 256, 0, stream>>>(state, index_devices,
                                                                  dev_cur, Dn);
        gemm(dev_cur, pWtgt, b_tgt, tgtb, Dn, 1024, 128, 0);
        fillf<<<blk((long long)Dn * 4), 256, 0, stream>>>(amax, -3.0e38f, Dn * 4);
        fillf<<<blk((long long)Dn * 4), 256, 0, stream>>>(wsum, 0.f, Dn * 4);
        fillf<<<blk((long long)Dn * 1024), 256, 0, stream>>>(msgD, 0.f, Dn * 1024);
        gat_alpha<<<F, 256, 0, stream>>>(srcb, tgtb, frag2dev_row, att, alphab, amax);
        gat_w<<<blk((long long)F * 4), 256, 0, stream>>>(alphab, amax, frag2dev_row,
                                                         wbuf, wsum, F);
        gat_msgd<<<blk((long long)F * 1024), 256, 0, stream>>>(srcb, wbuf, wsum,
                                                               frag2dev_row, msgD, F);
        gat_mean<<<blk((long long)Dn * 256), 256, 0, stream>>>(msgD, bias_att, msgDm, Dn);
        // --- device GRU ---
        gemm(msgDm, pDWih, D_bih, gi, Dn, 384, 256, 0);
        gemm(dev_cur, pDWhh, D_bhh, gh, Dn, 384, 128, 0);
        gru_pointwise<<<blk((long long)Dn * 128), 256, 0, stream>>>(
            gi, gh, dev_cur, dev_new, nullptr, nullptr, 0, Dn, 128, 0);
        // --- rebuild state ---
        fillf<<<blk((long long)nodes * 128), 256, 0, stream>>>(state, 0.f, nodes * 128);
        scatter_rows<<<blk((long long)Dn * 128), 256, 0, stream>>>(state, index_devices,
                                                                   dev_new, Dn);
        scatter_rows<<<blk((long long)F * 128), 256, 0, stream>>>(state, index_fragments,
                                                                  frag_new, F);
    }

    // ---- readout ----
    fillf<<<blk((long long)S * 128), 256, 0, stream>>>(pooled, 0.f, S * 128);
    pool_acc<<<blk((long long)F * 128), 256, 0, stream>>>(frag_new, frag_service, pooled, F);
    pool_scale<<<blk((long long)S * 128), 256, 0, stream>>>(pooled, lengths, S);
    // latency head -> dout[0:S]
    gemm(pooled, pWl1, bl1, mh1, S, 256, 128, 1);
    gemm(mh1, pWl2, bl2, mh2, S, 256, 256, 1);
    dot_out<<<S, 256, 0, stream>>>(mh2, Wl3, bl3, dout, 256);
    // throughput head -> dout[S:2S]
    gemm(h_srv, pWt1, bt1, mh1, S, 256, 128, 1);
    gemm(mh1, pWt2, bt2, mh2, S, 256, 256, 1);
    dot_out<<<S, 256, 0, stream>>>(mh2, Wt3, bt3, dout + S, 256);
    (void)n_in; (void)out_size; (void)ws_size;
}
